// Ours_44444321579629
// MI455X (gfx1250) — compile-verified, matrix operand order checked by reading, not run on hardware
//
#include <hip/hip_runtime.h>

typedef _Float16 f16;
typedef __attribute__((ext_vector_type(16))) _Float16 v16h;
typedef __attribute__((ext_vector_type(8)))  float    v8f;

union AFrag { v16h v; f16 h[16]; };

#define STR 68   // LDS row stride (floats): bank-conflict-free, rows 16B-aligned
#define WAVES 8

__device__ __forceinline__ v8f wmma16(const AFrag& a, const v16h& b, v8f c) {
  return __builtin_amdgcn_wmma_f32_16x16x32_f16(false, a.v, false, b, (short)0, c, false, false);
}

// Build A fragment from LDS stage; `row` = tileOff + l15
__device__ __forceinline__ void a_from_stage(const float* st, int row, int half, int kt, AFrag& a) {
#pragma unroll
  for (int run = 0; run < 2; ++run) {
    int kb = kt * 32 + run * 16 + half * 8;
    const float4* p = (const float4*)(st + row * STR + kb);
    float4 x0 = p[0], x1 = p[1];
    a.h[run*8+0] = (f16)x0.x; a.h[run*8+1] = (f16)x0.y;
    a.h[run*8+2] = (f16)x0.z; a.h[run*8+3] = (f16)x0.w;
    a.h[run*8+4] = (f16)x1.x; a.h[run*8+5] = (f16)x1.y;
    a.h[run*8+6] = (f16)x1.z; a.h[run*8+7] = (f16)x1.w;
  }
}

__device__ __forceinline__ void a_from_global(const float* src, AFrag& a, int run) {
  float4 x0 = ((const float4*)src)[0];
  float4 x1 = ((const float4*)src)[1];
  a.h[run*8+0] = (f16)x0.x; a.h[run*8+1] = (f16)x0.y;
  a.h[run*8+2] = (f16)x0.z; a.h[run*8+3] = (f16)x0.w;
  a.h[run*8+4] = (f16)x1.x; a.h[run*8+5] = (f16)x1.y;
  a.h[run*8+6] = (f16)x1.z; a.h[run*8+7] = (f16)x1.w;
}

// 64->64 layer, 2 M-tiles per wave, B fragments reused across tiles
__device__ __forceinline__ void layer64_mt2(const float* st, const f16* __restrict__ wp,
                                            const float* __restrict__ bias,
                                            int lane, int l15, int half, v8f c[2][4]) {
#pragma unroll
  for (int nt = 0; nt < 4; ++nt) {
    float bv = bias[nt * 16 + l15];
#pragma unroll
    for (int i = 0; i < 8; ++i) { c[0][nt][i] = bv; c[1][nt][i] = bv; }
  }
#pragma unroll
  for (int kt = 0; kt < 2; ++kt) {
    AFrag a0, a1;
    a_from_stage(st, l15,      half, kt, a0);
    a_from_stage(st, 16 + l15, half, kt, a1);
#pragma unroll
    for (int nt = 0; nt < 4; ++nt) {
      v16h b = *(const v16h*)(wp + (size_t)(kt * 4 + nt) * 512 + lane * 16);
      c[0][nt] = wmma16(a0, b, c[0][nt]);
      c[1][nt] = wmma16(a1, b, c[1][nt]);
    }
  }
}

// Write both tiles' D fragments (optionally ReLU'd) into LDS stage (D->row-major transpose)
__device__ __forceinline__ void stash_mt2(float* st, int l15, int half, v8f c[2][4], bool relu) {
#pragma unroll
  for (int t = 0; t < 2; ++t)
#pragma unroll
    for (int nt = 0; nt < 4; ++nt)
#pragma unroll
      for (int r = 0; r < 8; ++r) {
        float v = c[t][nt][r];
        if (relu) v = fmaxf(v, 0.f);
        st[(t * 16 + r + 8 * half) * STR + nt * 16 + l15] = v;
      }
  asm volatile("s_wait_dscnt 0x0" ::: "memory");
}

__device__ __forceinline__ void layernorm_mt2(v8f c[2][4], const float* __restrict__ g,
                                              const float* __restrict__ b, int l15) {
  float gv[4], bv[4];
#pragma unroll
  for (int nt = 0; nt < 4; ++nt) { gv[nt] = g[nt*16+l15]; bv[nt] = b[nt*16+l15]; }
#pragma unroll
  for (int t = 0; t < 2; ++t)
#pragma unroll
    for (int r = 0; r < 8; ++r) {
      float s = 0.f, s2 = 0.f;
#pragma unroll
      for (int nt = 0; nt < 4; ++nt) { float v = c[t][nt][r]; s += v; s2 += v * v; }
#pragma unroll
      for (int m = 1; m < 16; m <<= 1) { s += __shfl_xor(s, m, 32); s2 += __shfl_xor(s2, m, 32); }
      float mu  = s * (1.f / 64.f);
      float var = s2 * (1.f / 64.f) - mu * mu;
      float rs  = rsqrtf(var + 1e-5f);
#pragma unroll
      for (int nt = 0; nt < 4; ++nt) c[t][nt][r] = (c[t][nt][r] - mu) * rs * gv[nt] + bv[nt];
    }
}

// MODE 0: direct input [rows,K] -> store out0[rows,64]
// MODE 1: edge MLP (K=192): concat(nf[send], nf[recv], ef[e]) -> ef += y ; atomicAdd(agg[recv], y)
// MODE 2: node MLP (K=128): concat(nf[n], agg[n]) -> nf += y
template <int MODE>
__launch_bounds__(256)
__global__ void mlp3_kernel(const float* __restrict__ in0, const float* __restrict__ in1,
                            const int* __restrict__ conn,
                            const f16* __restrict__ w0, const f16* __restrict__ w1,
                            const f16* __restrict__ w2,
                            const float* __restrict__ b0, const float* __restrict__ b1,
                            const float* __restrict__ b2,
                            const float* __restrict__ gma, const float* __restrict__ bta,
                            float* __restrict__ out0, float* __restrict__ out1,
                            int rows, int K) {
  __shared__ float stage[WAVES][32 * STR];

  const int lane = threadIdx.x & 31;
  const int wv   = threadIdx.x >> 5;
  const int l15  = lane & 15;
  const int half = lane >> 4;
  const long rb0 = ((long)blockIdx.x * WAVES + wv) * 32;
  if (rb0 >= rows) return;                       // uniform per wave: EXEC stays all-ones
  const bool has2 = (rb0 + 16) < rows;           // N tail: second tile repeats tile 0, stores skipped
  const long rb1 = has2 ? rb0 + 16 : rb0;
  const int row0 = (int)rb0 + l15;
  const int row1 = (int)rb1 + l15;
  float* st = stage[wv];

  int sI[2] = {0, 0}, rI[2] = {0, 0};
  if (MODE == 1) {
    sI[0] = conn[2 * (size_t)row0];     rI[0] = conn[2 * (size_t)row0 + 1];
    sI[1] = conn[2 * (size_t)row1];     rI[1] = conn[2 * (size_t)row1 + 1];
  }

  v8f c[2][4];
  // ---- layer 0 ----
#pragma unroll
  for (int nt = 0; nt < 4; ++nt) {
    float bv = b0[nt * 16 + l15];
#pragma unroll
    for (int i = 0; i < 8; ++i) { c[0][nt][i] = bv; c[1][nt][i] = bv; }
  }
  const int nkt = (K + 31) >> 5;
  for (int kt = 0; kt < nkt; ++kt) {
    AFrag a[2];
#pragma unroll
    for (int t = 0; t < 2; ++t) {
      int myRow = t ? row1 : row0;
#pragma unroll
      for (int run = 0; run < 2; ++run) {
        int kb = kt * 32 + run * 16 + half * 8;
        if (MODE == 0) {
#pragma unroll
          for (int j = 0; j < 8; ++j) {
            int k = kb + j;
            a[t].h[run * 8 + j] = (f16)((k < K) ? in0[(size_t)myRow * K + k] : 0.f);
          }
        } else {
          const float* src;
          if (MODE == 1) {
            if (kb < 64)       src = in0 + (size_t)sI[t] * 64 + kb;
            else if (kb < 128) src = in0 + (size_t)rI[t] * 64 + (kb - 64);
            else               src = in1 + (size_t)myRow * 64 + (kb - 128);
          } else {
            src = (kb < 64) ? in0 + (size_t)myRow * 64 + kb
                            : in1 + (size_t)myRow * 64 + (kb - 64);
          }
          a_from_global(src, a[t], run);
        }
      }
    }
#pragma unroll
    for (int nt = 0; nt < 4; ++nt) {
      v16h b = *(const v16h*)(w0 + (size_t)(kt * 4 + nt) * 512 + lane * 16);
      c[0][nt] = wmma16(a[0], b, c[0][nt]);
      c[1][nt] = wmma16(a[1], b, c[1][nt]);
    }
  }
  stash_mt2(st, l15, half, c, true);
  layer64_mt2(st, w1, b1, lane, l15, half, c);
  stash_mt2(st, l15, half, c, true);
  layer64_mt2(st, w2, b2, lane, l15, half, c);
  layernorm_mt2(c, gma, bta, l15);
  stash_mt2(st, l15, half, c, false);

  // ---- vectorized output: each lane owns one row-half (32 contiguous cols) per tile ----
#pragma unroll
  for (int t = 0; t < 2; ++t) {
    if (t == 1 && !has2) break;                  // uniform
    size_t row = (size_t)((t ? rb1 : rb0) + l15);
    const float* srow = st + (t * 16 + l15) * STR + half * 32;
    if (MODE == 0) {
#pragma unroll
      for (int j = 0; j < 8; ++j)
        ((float4*)(out0 + row * 64 + half * 32))[j] = ((const float4*)srow)[j];
    } else if (MODE == 1) {
      int rv = rI[t];
      float* erow = out0 + row * 64 + half * 32;
      float* arow = out1 + (size_t)rv * 64 + half * 32;
#pragma unroll
      for (int j = 0; j < 8; ++j) {
        float4 v = ((const float4*)srow)[j];
        float4 e = ((const float4*)erow)[j];
        e.x += v.x; e.y += v.y; e.z += v.z; e.w += v.w;
        ((float4*)erow)[j] = e;                  // ef residual
        atomicAdd(arow + 4 * j + 0, v.x);        // segment-sum scatter
        atomicAdd(arow + 4 * j + 1, v.y);
        atomicAdd(arow + 4 * j + 2, v.z);
        atomicAdd(arow + 4 * j + 3, v.w);
      }
    } else {
      float* nrow = out0 + row * 64 + half * 32;
#pragma unroll
      for (int j = 0; j < 8; ++j) {
        float4 v = ((const float4*)srow)[j];
        float4 n = ((const float4*)nrow)[j];
        n.x += v.x; n.y += v.y; n.z += v.z; n.w += v.w;
        ((float4*)nrow)[j] = n;                  // nf residual
      }
    }
  }
}

// Decoder: 64->64 relu, 64->64 relu (WMMA), 64->3 scalar; 16 rows/wave (N % 16 == 0)
__launch_bounds__(256)
__global__ void dec_kernel(const float* __restrict__ nf,
                           const f16* __restrict__ w0, const f16* __restrict__ w1,
                           const float* __restrict__ b0, const float* __restrict__ b1,
                           const float* __restrict__ w2f, const float* __restrict__ b2,
                           const float* __restrict__ omean, const float* __restrict__ ostd,
                           float* __restrict__ outHat, float* __restrict__ outNoh, int rows) {
  __shared__ float stage[WAVES][16 * STR];
  const int lane = threadIdx.x & 31;
  const int wv   = threadIdx.x >> 5;
  const int l15  = lane & 15;
  const int half = lane >> 4;
  const long rowBase = ((long)blockIdx.x * WAVES + wv) * 16;
  if (rowBase >= rows) return;
  const int myRow = (int)rowBase + l15;
  float* st = stage[wv];

  v8f c[4];
#pragma unroll
  for (int nt = 0; nt < 4; ++nt) {
    float bv = b0[nt * 16 + l15];
#pragma unroll
    for (int i = 0; i < 8; ++i) c[nt][i] = bv;
  }
#pragma unroll
  for (int kt = 0; kt < 2; ++kt) {
    AFrag a;
#pragma unroll
    for (int run = 0; run < 2; ++run) {
      int kb = kt * 32 + run * 16 + half * 8;
      a_from_global(nf + (size_t)myRow * 64 + kb, a, run);
    }
#pragma unroll
    for (int nt = 0; nt < 4; ++nt) {
      v16h b = *(const v16h*)(w0 + (size_t)(kt * 4 + nt) * 512 + lane * 16);
      c[nt] = wmma16(a, b, c[nt]);
    }
  }
  // stash + layer 1 (single tile)
#pragma unroll
  for (int nt = 0; nt < 4; ++nt)
#pragma unroll
    for (int r = 0; r < 8; ++r)
      st[(r + 8 * half) * STR + nt * 16 + l15] = fmaxf(c[nt][r], 0.f);
  asm volatile("s_wait_dscnt 0x0" ::: "memory");

#pragma unroll
  for (int nt = 0; nt < 4; ++nt) {
    float bv = b1[nt * 16 + l15];
#pragma unroll
    for (int i = 0; i < 8; ++i) c[nt][i] = bv;
  }
#pragma unroll
  for (int kt = 0; kt < 2; ++kt) {
    AFrag a; a_from_stage(st, l15, half, kt, a);
#pragma unroll
    for (int nt = 0; nt < 4; ++nt) {
      v16h b = *(const v16h*)(w1 + (size_t)(kt * 4 + nt) * 512 + lane * 16);
      c[nt] = wmma16(a, b, c[nt]);
    }
  }
#pragma unroll
  for (int nt = 0; nt < 4; ++nt)
#pragma unroll
    for (int r = 0; r < 8; ++r)
      st[(r + 8 * half) * STR + nt * 16 + l15] = fmaxf(c[nt][r], 0.f);
  asm volatile("s_wait_dscnt 0x0" ::: "memory");

  if (lane < 16) {
    size_t node = (size_t)rowBase + l15;
#pragma unroll
    for (int j = 0; j < 3; ++j) {
      float acc = b2[j];
      for (int k = 0; k < 64; ++k) acc += st[l15 * STR + k] * w2f[k * 3 + j];
      outNoh[node * 3 + j] = acc;
      outHat[node * 3 + j] = acc * ostd[j] + omean[j];
    }
  }
}

// Pack f32 weights [K][64] into f16 B-fragment layout: [kt][nt][lane][16]
__global__ void pack_kernel(const float* __restrict__ W, int K, f16* __restrict__ dst, int total) {
  int t = blockIdx.x * 256 + threadIdx.x;
  if (t >= total) return;
  int e    = t & 15;
  int lane = (t >> 4) & 31;
  int frag = t >> 9;
  int nt   = frag & 3;
  int kt   = frag >> 2;
  int n    = nt * 16 + (lane & 15);
  int half = lane >> 4;
  int k    = kt * 32 + ((e < 8) ? (half * 8 + e) : (16 + half * 8 + (e - 8)));
  float v  = (k < K) ? W[(size_t)k * 64 + n] : 0.f;
  dst[t] = (f16)v;
}

__global__ void nodefeat_kernel(const float* __restrict__ node, const float* __restrict__ pos,
                                const float* __restrict__ m, const float* __restrict__ s,
                                float* __restrict__ nf_in, int N) {
  int n = blockIdx.x * 256 + threadIdx.x;
  if (n >= N) return;
#pragma unroll
  for (int i = 0; i < 9; ++i) {
    float v = (i < 6) ? node[(size_t)n * 6 + i] : pos[(size_t)n * 3 + (i - 6)];
    nf_in[(size_t)n * 9 + i] = (v - m[i]) / s[i];
  }
}

__global__ void edgefeat_kernel(const float* __restrict__ pos, const int* __restrict__ conn,
                                const float* __restrict__ m, const float* __restrict__ s,
                                float* __restrict__ ef_in, int E) {
  int e = blockIdx.x * 256 + threadIdx.x;
  if (e >= E) return;
  int si = conn[2 * (size_t)e], ri = conn[2 * (size_t)e + 1];
  float dx = pos[3 * (size_t)ri + 0] - pos[3 * (size_t)si + 0];
  float dy = pos[3 * (size_t)ri + 1] - pos[3 * (size_t)si + 1];
  float dz = pos[3 * (size_t)ri + 2] - pos[3 * (size_t)si + 2];
  float d  = sqrtf(dx * dx + dy * dy + dz * dz);
  ef_in[4 * (size_t)e + 0] = (dx - m[0]) / s[0];
  ef_in[4 * (size_t)e + 1] = (dy - m[1]) / s[1];
  ef_in[4 * (size_t)e + 2] = (dz - m[2]) / s[2];
  ef_in[4 * (size_t)e + 3] = (d  - m[3]) / s[3];
}

__global__ void normout_kernel(const float* __restrict__ outp, const float* __restrict__ m,
                               const float* __restrict__ s, float* __restrict__ dst, int total) {
  int i = blockIdx.x * 256 + threadIdx.x;
  if (i >= total) return;
  int c = i % 3;
  dst[i] = (outp[i] - m[c]) / s[c];
}

extern "C" void kernel_launch(void* const* d_in, const int* in_sizes, int n_in,
                              void* d_out, int out_size, void* d_ws, size_t ws_size,
                              hipStream_t stream) {
  (void)n_in; (void)out_size; (void)ws_size;
  const float* pos     = (const float*)d_in[0];
  const float* node    = (const float*)d_in[1];
  const int*   conn    = (const int*)d_in[2];
  const float* outp    = (const float*)d_in[3];
  const float* nn_mean = (const float*)d_in[4];
  const float* nn_std  = (const float*)d_in[5];
  const float* ne_mean = (const float*)d_in[6];
  const float* ne_std  = (const float*)d_in[7];
  const float* no_mean = (const float*)d_in[8];
  const float* no_std  = (const float*)d_in[9];

  const int N = in_sizes[0] / 3;
  const int E = in_sizes[2] / 2;

  // ---- packed weight table: enc_node(3), enc_edge(3), 3x[edge(3)+node(3)], dec(2) ----
  const int pidx[26] = {10,12,14, 18,20,22, 26,28,30, 34,36,38, 42,44,46,
                        50,52,54, 58,60,62, 66,68,70, 74,76};
  const int pK[26]   = {9,64,64, 4,64,64, 192,64,64, 128,64,64, 192,64,64,
                        128,64,64, 192,64,64, 128,64,64, 64,64};
  size_t woff[26], off = 0;
  for (int i = 0; i < 26; ++i) { woff[i] = off; off += (size_t)((pK[i] + 31) & ~31) * 64; }

  char* ws   = (char*)d_ws;
  f16*  wpck = (f16*)ws;
  size_t cur = ((off * sizeof(f16)) + 255) & ~(size_t)255;
  float* nf    = (float*)(ws + cur); cur += (size_t)N * 64 * 4;
  float* ef    = (float*)(ws + cur); cur += (size_t)E * 64 * 4;
  float* agg   = (float*)(ws + cur); cur += (size_t)N * 64 * 4;
  float* nf_in = (float*)(ws + cur); cur += (size_t)N * 9 * 4;
  float* ef_in = (float*)(ws + cur); cur += (size_t)E * 4 * 4;

  // weight packing (cheap; weights stay L2-resident afterwards)
  for (int i = 0; i < 26; ++i) {
    int total = ((pK[i] + 31) & ~31) * 64;
    pack_kernel<<<(total + 255) / 256, 256, 0, stream>>>((const float*)d_in[pidx[i]], pK[i],
                                                         wpck + woff[i], total);
  }

  nodefeat_kernel<<<(N + 255) / 256, 256, 0, stream>>>(node, pos, nn_mean, nn_std, nf_in, N);
  edgefeat_kernel<<<(E + 255) / 256, 256, 0, stream>>>(pos, conn, ne_mean, ne_std, ef_in, E);

  // encoders (256 threads = 8 waves; 32 rows/wave -> 256 rows/block)
  mlp3_kernel<0><<<(N + 255) / 256, 256, 0, stream>>>(
      nf_in, nullptr, nullptr, wpck + woff[0], wpck + woff[1], wpck + woff[2],
      (const float*)d_in[11], (const float*)d_in[13], (const float*)d_in[15],
      (const float*)d_in[16], (const float*)d_in[17], nf, nullptr, N, 9);
  mlp3_kernel<0><<<(E + 255) / 256, 256, 0, stream>>>(
      ef_in, nullptr, nullptr, wpck + woff[3], wpck + woff[4], wpck + woff[5],
      (const float*)d_in[19], (const float*)d_in[21], (const float*)d_in[23],
      (const float*)d_in[24], (const float*)d_in[25], ef, nullptr, E, 4);

  // message-passing blocks
  for (int blk = 0; blk < 3; ++blk) {
    int eb = 26 + 16 * blk;        // edge MLP leaf base
    int nb = eb + 8;               // node MLP leaf base
    int wE = 6 + 6 * blk;          // packed index of edge W0
    int wN = wE + 3;
    hipMemsetAsync(agg, 0, (size_t)N * 64 * 4, stream);
    mlp3_kernel<1><<<(E + 255) / 256, 256, 0, stream>>>(
        nf, ef, conn, wpck + woff[wE], wpck + woff[wE + 1], wpck + woff[wE + 2],
        (const float*)d_in[eb + 1], (const float*)d_in[eb + 3], (const float*)d_in[eb + 5],
        (const float*)d_in[eb + 6], (const float*)d_in[eb + 7], ef, agg, E, 192);
    mlp3_kernel<2><<<(N + 255) / 256, 256, 0, stream>>>(
        nf, agg, nullptr, wpck + woff[wN], wpck + woff[wN + 1], wpck + woff[wN + 2],
        (const float*)d_in[nb + 1], (const float*)d_in[nb + 3], (const float*)d_in[nb + 5],
        (const float*)d_in[nb + 6], (const float*)d_in[nb + 7], nf, nullptr, N, 128);
  }

  // decoder -> output_hat (slot 0) and normalized_output_hat (slot 2)
  float* out_hat = (float*)d_out;
  float* out_no  = out_hat + (size_t)N * 3;
  float* out_noh = out_hat + (size_t)2 * N * 3;
  dec_kernel<<<(N + 127) / 128, 256, 0, stream>>>(
      nf, wpck + woff[24], wpck + woff[25],
      (const float*)d_in[75], (const float*)d_in[77],
      (const float*)d_in[78], (const float*)d_in[79],
      no_mean, no_std, out_hat, out_noh, N);

  // normalized_output (slot 1)
  normout_kernel<<<(N * 3 + 255) / 256, 256, 0, stream>>>(outp, no_mean, no_std, out_no, N * 3);
}
